// UniCoreLinearFP8_8504035246561
// MI455X (gfx1250) — 8-bit matrix operand order repaired and checked by reading, compiled, not run on hardware
//
#include <hip/hip_runtime.h>
#include <stdint.h>


typedef int   v2i  __attribute__((ext_vector_type(2)));
typedef int   v4i  __attribute__((ext_vector_type(4)));
typedef int   v16i __attribute__((ext_vector_type(16)));
typedef float v4f  __attribute__((ext_vector_type(4)));
typedef float v8f  __attribute__((ext_vector_type(8)));

#define TILE_M 128
#define TILE_N 128
#define TILE_K 128
#define LDS_PITCH 144

#define AS1 __attribute__((address_space(1)))
#define AS3 __attribute__((address_space(3)))

#if __has_builtin(__builtin_amdgcn_global_load_async_to_lds_b128)
#define HAVE_ASYNC_LDS 1
#else
#define HAVE_ASYNC_LDS 0
#endif


__device__ __forceinline__ unsigned short f32_to_bf16_rne(float f) {
    unsigned int u = __float_as_uint(f);
    unsigned int lsb = (u >> 16) & 1u;
    u += 0x7FFFu + lsb;
    return (unsigned short)(u >> 16);
}

__device__ __forceinline__ unsigned int pack_e4m3(float x) {
    const unsigned short b = f32_to_bf16_rne(x);
    const unsigned int sign = ((unsigned int)(b & 0x8000u)) >> 8;
    const unsigned int mag  = b & 0x7FFFu;
    if (mag == 0u) return sign;
    const float absw = __uint_as_float(mag << 16);
    if (absw < 0.015625f) {
        int mi = (int)rintf(absw * 512.0f);
        if (mi > 7) mi = 7;
        return sign | (unsigned int)mi;
    }
    int e = (int)((b >> 7) & 0xFF) - 127;
    e = e < -6 ? -6 : (e > 8 ? 8 : e);
    const unsigned int e4 = (unsigned int)(((e + 7) & 15) << 3);
    const unsigned int m3 = (unsigned int)((b >> 4) & 7);
    return sign | e4 | m3;
}

__global__ void __launch_bounds__(256)
quant_e4m3_kernel(const float* __restrict__ in,
                  unsigned int* __restrict__ out,
                  long long nvec) {
    const long long i = (long long)blockIdx.x * blockDim.x + threadIdx.x;
    if (i >= nvec) return;
    const v4f v = __builtin_nontemporal_load((const v4f*)in + i);
    const unsigned int w = pack_e4m3(v.x)
                         | (pack_e4m3(v.y) << 8)
                         | (pack_e4m3(v.z) << 16)
                         | (pack_e4m3(v.w) << 24);
    out[i] = w;
}


__global__ void __launch_bounds__(256)
fp8_gemm_kernel(const unsigned char* __restrict__ Aq,
                const unsigned char* __restrict__ Bq,
                const float* __restrict__ xs,
                const float* __restrict__ wsc,
                float* __restrict__ out,
                int M, int N, int K) {
    __shared__ __align__(16) unsigned char lsA[2][TILE_M * LDS_PITCH];
    __shared__ __align__(16) unsigned char lsB[2][TILE_N * LDS_PITCH];

    const int tid  = threadIdx.x;
    const int lane = tid & 31;
    const int wv   = tid >> 5;
    const int wm   = (wv & 3) * 32;
    const int wn   = (wv >> 2) * 64;
    const int lr   = lane & 15;
    const int hk8  = (lane >> 4) * 8;
    const int hk16 = (lane >> 4) * 16;

    const int m0 = blockIdx.y * TILE_M;
    const int n0 = blockIdx.x * TILE_N;

    const int ldrow = tid >> 1;
    const int ldseg = (tid & 1) * 64;
    const unsigned char* gA = Aq + (size_t)(m0 + ldrow) * K + ldseg;
    const unsigned char* gB = Bq + (size_t)(n0 + ldrow) * K + ldseg;
    const int sOff = ldrow * LDS_PITCH + ldseg;

    auto issue_tile = [&](int k0, int buf) {
        unsigned char* sa = &lsA[buf][sOff];
        unsigned char* sb = &lsB[buf][sOff];
#if HAVE_ASYNC_LDS
#pragma unroll
        for (int i = 0; i < 4; ++i)
            __builtin_amdgcn_global_load_async_to_lds_b128(
                (AS1 v4i*)(gA + k0 + i * 16), (AS3 v4i*)(sa + i * 16), 0, 0);
#pragma unroll
        for (int i = 0; i < 4; ++i)
            __builtin_amdgcn_global_load_async_to_lds_b128(
                (AS1 v4i*)(gB + k0 + i * 16), (AS3 v4i*)(sb + i * 16), 0, 0);
#else
#pragma unroll
        for (int i = 0; i < 4; ++i)
            *(v4i*)(sa + i * 16) = *(const v4i*)(gA + k0 + i * 16);
#pragma unroll
        for (int i = 0; i < 4; ++i)
            *(v4i*)(sb + i * 16) = *(const v4i*)(gB + k0 + i * 16);
#endif
    };

    const v8f vzero = {0.f, 0.f, 0.f, 0.f, 0.f, 0.f, 0.f, 0.f};
    v8f acc[2][4];
#pragma unroll
    for (int fm = 0; fm < 2; ++fm)
#pragma unroll
        for (int fn = 0; fn < 4; ++fn) acc[fm][fn] = vzero;

    const int nIter = K / TILE_K;
    issue_tile(0, 0);

    for (int it = 0; it < nIter; ++it) {
        const int cur = it & 1;
#if HAVE_ASYNC_LDS
#if __has_builtin(__builtin_amdgcn_s_wait_asynccnt)
        __builtin_amdgcn_s_wait_asynccnt(0);
#else
        asm volatile("s_wait_asynccnt 0" ::: "memory");
#endif
#endif
        __syncthreads();

        if (it + 1 < nIter) issue_tile((it + 1) * TILE_K, cur ^ 1);

        v16i afrag[2];
#pragma unroll
        for (int fm = 0; fm < 2; ++fm) {
            const unsigned char* ab =
                &lsA[cur][(wm + fm * 16 + lr) * LDS_PITCH + hk16];
#pragma unroll
            for (int j = 0; j < 8; ++j) {
                v2i t = *(const v2i*)(ab + (j >> 1) * 32 + (j & 1) * 8);
                afrag[fm][2 * j]     = t.x;
                afrag[fm][2 * j + 1] = t.y;
            }
        }

        v16i bfrag[4];
#pragma unroll
        for (int fn = 0; fn < 4; ++fn) {
            const unsigned char* bb =
                &lsB[cur][(wn + fn * 16 + lr) * LDS_PITCH + hk16];
#pragma unroll
            for (int g = 0; g < 4; ++g) {
                v4i t = *(const v4i*)(bb + g * 32);
                bfrag[fn][4 * g]     = t.x;
                bfrag[fn][4 * g + 1] = t.y;
                bfrag[fn][4 * g + 2] = t.z;
                bfrag[fn][4 * g + 3] = t.w;
            }
        }

#pragma unroll
        for (int fm = 0; fm < 2; ++fm)
#pragma unroll
            for (int fn = 0; fn < 4; ++fn)
                acc[fm][fn] = __builtin_amdgcn_wmma_f32_16x16x128_fp8_fp8(
                    afrag[fm], bfrag[fn], (short)0, acc[fm][fn],
                     false,  false);
    }

    const int colBase = n0 + wn + lr;
    float wsv[4];
#pragma unroll
    for (int fn = 0; fn < 4; ++fn) wsv[fn] = wsc[colBase + fn * 16];

#pragma unroll
    for (int fm = 0; fm < 2; ++fm) {
        const int mBase = m0 + wm + fm * 16 + (lane >> 4) * 8;
#pragma unroll
        for (int r = 0; r < 8; ++r) {
            const int m = mBase + r;
            const float s = xs[m];
            float* orow = out + (size_t)m * N + colBase;
#pragma unroll
            for (int fn = 0; fn < 4; ++fn)
                __builtin_nontemporal_store(acc[fm][fn][r] * s * wsv[fn],
                                            orow + fn * 16);
        }
    }
}


extern "C" void kernel_launch(void* const* d_in, const int* in_sizes, int n_in,
                              void* d_out, int out_size, void* d_ws,
                              size_t ws_size, hipStream_t stream) {
    const float* x_q     = (const float*)d_in[0];
    const float* x_scale = (const float*)d_in[1];
    const float* weight  = (const float*)d_in[2];
    const float* wscales = (const float*)d_in[3];
    float* out = (float*)d_out;

    const long long MK = (long long)in_sizes[0];
    const int M = in_sizes[1];
    const long long NK = (long long)in_sizes[2];
    const int N = in_sizes[3];
    const int K = (int)(MK / (long long)M);

    unsigned char* xq8 = (unsigned char*)d_ws;
    const size_t xq_bytes = ((size_t)MK + 255u) & ~(size_t)255u;
    unsigned char* wq8 = xq8 + xq_bytes;

    {
        const long long nv = MK / 4;
        const unsigned int blocks = (unsigned int)((nv + 255) / 256);
        quant_e4m3_kernel<<<dim3(blocks), dim3(256), 0, stream>>>(
            x_q, (unsigned int*)xq8, nv);
    }
    {
        const long long nv = NK / 4;
        const unsigned int blocks = (unsigned int)((nv + 255) / 256);
        quant_e4m3_kernel<<<dim3(blocks), dim3(256), 0, stream>>>(
            weight, (unsigned int*)wq8, nv);
    }

    dim3 grid((unsigned)(N / TILE_N), (unsigned)(M / TILE_M));
    fp8_gemm_kernel<<<grid, dim3(256), 0, stream>>>(
        xq8, wq8, x_scale, wscales, out, M, N, K);
}
